// PSMM_80977313398785
// MI455X (gfx1250) — matrix operand order repaired and checked by reading, not compiled, not run on hardware
//
#include <hip/hip_runtime.h>
#include <hip/hip_bf16.h>
#include <cmath>

// ---------------- WMMA types & fragment loaders (gfx1250, wave32) ----------
typedef __attribute__((ext_vector_type(16))) _Float16 v16h;
typedef __attribute__((ext_vector_type(8)))  float    v8f;
typedef int v4i_ld __attribute__((vector_size(16)));   // matches builtin param

union F16x16 { v16h v; uint4 q[2]; };

// A fragment: 16(M) x 32(K), A row-major [m][k], lda in elements (mult of 8).
// Lane l: m = m0 + (l&15), hi = l>>4.
// dwords 0..3 -> K = kb + hi*8 + {0..7};  dwords 4..7 -> K = kb + 16 + hi*8 + {0..7}
__device__ inline v16h load_frag_a(const _Float16* __restrict__ A, int lda,
                                   int m0, int kb, int lane) {
    int m = m0 + (lane & 15);
    int hi = lane >> 4;
    const _Float16* p = A + (long long)m * lda + kb + hi * 8;
    F16x16 f;
    f.q[0] = *(const uint4*)p;
    f.q[1] = *(const uint4*)(p + 16);
    return f.v;
}

// B fragment: 32(K) x 16(N), computing A@B^T with B stored row-major [n][k].
// Lane l: n = n0 + (l&15), hi = l>>4; K = kb + hi*8 + {0..7}, then kb + 16 + hi*8 + {0..7}.
__device__ inline v16h load_frag_b(const _Float16* __restrict__ B, int ldb,
                                   int n0, int kb, int lane) {
    int n = n0 + (lane & 15);
    int hi = lane >> 4;
    const _Float16* p = B + (long long)n * ldb + kb + hi * 8;
    F16x16 f;
    f.q[0] = *(const uint4*)p;
    f.q[1] = *(const uint4*)(p + 16);
    return f.v;
}

__device__ inline v8f wmma_step(v16h a, v16h b, v8f c) {
    return __builtin_amdgcn_wmma_f32_16x16x32_f16(false, a, false, b,
                                                  (short)0, c, false, false);
}

// async global->LDS staging helper (16B per lane-op), with portable fallback
__device__ inline void stage_to_lds_16B(const void* g, void* l) {
#if defined(__has_builtin) && __has_builtin(__builtin_amdgcn_global_load_async_to_lds_b128)
    __builtin_amdgcn_global_load_async_to_lds_b128(
        (__attribute__((address_space(1))) v4i_ld*)g,
        (__attribute__((address_space(3))) v4i_ld*)l, 0, 0);
#else
    *(uint4*)l = *(const uint4*)g;
#endif
}
__device__ inline void wait_async_done() {
#if defined(__has_builtin) && __has_builtin(__builtin_amdgcn_global_load_async_to_lds_b128)
#if __has_builtin(__builtin_amdgcn_s_wait_asynccnt)
    __builtin_amdgcn_s_wait_asynccnt(0);
#else
    asm volatile("s_wait_asynccnt 0" ::: "memory");
#endif
#endif
}

// ---------------- generic batched WMMA GEMM: C = A @ B^T (+bias) (opt tanh) -
// A: f16 [M x K] row stride lda;  B: f16 [N x K] row stride ldb
// Block 256 thr = 8 waves (4 M-waves x 2 N-waves); block tile 128(M) x 64(N);
// wave tile 32x32 = 2x2 fragments. Requires M%128==0, K%32==0, N%16==0.
__global__ __launch_bounds__(256) void gemm_wmma(
    const _Float16* __restrict__ A, const _Float16* __restrict__ B,
    const float* __restrict__ bias1, const float* __restrict__ bias2,
    float* __restrict__ C32, _Float16* __restrict__ C16,
    int M, int N, int K, int lda, int ldb, int ldc,
    long long sA, long long sB, long long sC, int act /*1=tanh*/)
{
    int z = blockIdx.z;
    A += (long long)z * sA;
    B += (long long)z * sB;
    int wid  = threadIdx.x >> 5, lane = threadIdx.x & 31;
    int m0 = blockIdx.y * 128 + (wid & 3) * 32;
    int n0 = blockIdx.x * 64  + (wid >> 2) * 32;
    int hi = lane >> 4, lo = lane & 15;

    v8f acc[2][2] = {};
    bool vn0 = (n0 < N), vn1 = (n0 + 16 < N);

    // lane-private row bases for prefetching the next K tile
    const _Float16* pa = A + (long long)(m0 + lo) * lda;
    const _Float16* pb = B + (long long)((vn0 ? n0 : 0) + lo) * ldb;

    if (vn1) {                       // full 2x2 fast path (no guards in loop)
        for (int kb = 0; kb < K; kb += 32) {
            v16h a0 = load_frag_a(A, lda, m0,      kb, lane);
            v16h a1 = load_frag_a(A, lda, m0 + 16, kb, lane);
            v16h b0 = load_frag_b(B, ldb, n0,      kb, lane);
            v16h b1 = load_frag_b(B, ldb, n0 + 16, kb, lane);
            if (kb + 32 < K) {
                __builtin_prefetch(pa + kb + 32, 0, 1);
                __builtin_prefetch(pb + kb + 32, 0, 1);
            }
            acc[0][0] = wmma_step(a0, b0, acc[0][0]);
            acc[0][1] = wmma_step(a0, b1, acc[0][1]);
            acc[1][0] = wmma_step(a1, b0, acc[1][0]);
            acc[1][1] = wmma_step(a1, b1, acc[1][1]);
        }
    } else if (vn0) {                // N-edge: single valid B fragment
        for (int kb = 0; kb < K; kb += 32) {
            v16h a0 = load_frag_a(A, lda, m0,      kb, lane);
            v16h a1 = load_frag_a(A, lda, m0 + 16, kb, lane);
            v16h b0 = load_frag_b(B, ldb, n0,      kb, lane);
            acc[0][0] = wmma_step(a0, b0, acc[0][0]);
            acc[1][0] = wmma_step(a1, b0, acc[1][0]);
        }
    }

    #pragma unroll
    for (int fm = 0; fm < 2; fm++) {
        #pragma unroll
        for (int fn = 0; fn < 2; fn++) {
            int nn0 = n0 + fn * 16;
            if (nn0 >= N) continue;
            int col = nn0 + lo;
            float bs = (bias1 ? bias1[col] : 0.f) + (bias2 ? bias2[col] : 0.f);
            #pragma unroll
            for (int r = 0; r < 8; r++) {
                int row = m0 + fm * 16 + hi * 8 + r;
                float v = acc[fm][fn][r] + bs;
                if (act) v = tanhf(v);
                long long ci = (long long)z * sC + (long long)row * ldc + col;
                if (C32) C32[ci] = v;
                if (C16) C16[ci] = (_Float16)v;
            }
        }
    }
}

// ---------------- small kernels --------------------------------------------
__global__ void cast_f32_f16(const float* __restrict__ in,
                             _Float16* __restrict__ out, int n) {
    int i = blockIdx.x * 256 + threadIdx.x;
    if (i < n) out[i] = (_Float16)in[i];
}

__global__ void embed_kernel(const int* __restrict__ ids,
                             const float* __restrict__ embed_w,
                             _Float16* __restrict__ X16) {
    int i = blockIdx.x * 256 + threadIdx.x;        // 4096*512 total
    int r = i >> 9, d = i & 511;
    int tok = ids[r];
    X16[i] = (_Float16)embed_w[(long long)tok * 512 + d];
}

// gatesH[32x2048] = h16[32x512] @ Whh16^T + gatesX[t]
// h16 (32KB) is staged into LDS once per block via async global->LDS copy;
// every wave's WMMA A-fragments then come from LDS (ds_load_b128).
__global__ __launch_bounds__(256) void lstm_step_matmul(
    const _Float16* __restrict__ h16, const _Float16* __restrict__ Whh16,
    const float* __restrict__ gatesX, float* __restrict__ gatesH, int t)
{
    __shared__ __align__(16) _Float16 hl[32 * 512];   // 32 KB
    int tid = threadIdx.x;
    // 32768 B / 16 B = 2048 chunks, 8 per thread
    for (int c = tid; c < 2048; c += 256)
        stage_to_lds_16B((const char*)h16 + (size_t)c * 16,
                         (char*)hl + (size_t)c * 16);
    wait_async_done();
    __syncthreads();

    int wid = tid >> 5, lane = tid & 31;
    int w = blockIdx.x * 8 + wid;                  // 0..255 tiles
    int m0 = (w & 1) * 16;
    int n0 = (w >> 1) * 16;
    int hi = lane >> 4, lo = lane & 15;
    const float* gx = gatesX + (long long)(t * 32) * 2048;

    v8f acc;
    #pragma unroll
    for (int r = 0; r < 8; r++)
        acc[r] = gx[(long long)(m0 + hi * 8 + r) * 2048 + n0 + lo];

    for (int kb = 0; kb < 512; kb += 32) {
        v16h a = load_frag_a(hl,    512, m0, kb, lane);   // from LDS
        v16h b = load_frag_b(Whh16, 512, n0, kb, lane);   // from L2
        acc = wmma_step(a, b, acc);
    }
    #pragma unroll
    for (int r = 0; r < 8; r++)
        gatesH[(long long)(m0 + hi * 8 + r) * 2048 + n0 + lo] = acc[r];
}

__global__ void lstm_update(const float* __restrict__ gatesH,
                            float* __restrict__ c32, _Float16* __restrict__ h16,
                            _Float16* __restrict__ H16, int t)
{
    int idx = blockIdx.x * 256 + threadIdx.x;      // 0..16383
    int bb = idx >> 9, d = idx & 511;
    const float* g = gatesH + (long long)bb * 2048;
    float gi = g[d], gf = g[512 + d], gg = g[1024 + d], go = g[1536 + d];
    float si = 1.f / (1.f + __expf(-gi));
    float sf = 1.f / (1.f + __expf(-gf));
    float so = 1.f / (1.f + __expf(-go));
    float c = sf * c32[idx] + si * tanhf(gg);
    float h = so * tanhf(c);
    c32[idx] = c;
    _Float16 hh = (_Float16)h;
    h16[idx] = hh;
    H16[(long long)(t * 32 + bb) * 512 + d] = hh;
}

// sent[r] = dot(Q16[r,:], sentinel)   one wave32 per row
__global__ void sent_kernel(const _Float16* __restrict__ Q16,
                            const float* __restrict__ sentinel,
                            float* __restrict__ sent)
{
    int wid = threadIdx.x >> 5, lane = threadIdx.x & 31;
    int row = blockIdx.x * 8 + wid;
    float s = 0.f;
    for (int d = lane; d < 512; d += 32)
        s += (float)Q16[(long long)row * 512 + d] * sentinel[d];
    for (int off = 16; off; off >>= 1) s += __shfl_down(s, off, 32);
    if (lane == 0) sent[row] = s;
}

// causal softmax over (scores row + sentinel); block = 128 thr per (t,b)
__global__ void attn_softmax(const float* __restrict__ S,
                             const float* __restrict__ sent,
                             float* __restrict__ a, float* __restrict__ gsent)
{
    __shared__ float red[128];
    int r = blockIdx.x, t = r >> 5, b = r & 31;
    int j = threadIdx.x;
    long long base = (long long)b * 16384 + t * 128;
    float s = (j <= t) ? S[base + j] : -INFINITY;
    float ssent = sent[r];
    red[j] = s; __syncthreads();
    for (int off = 64; off; off >>= 1) {
        if (j < off) red[j] = fmaxf(red[j], red[j + off]);
        __syncthreads();
    }
    float mx = fmaxf(red[0], ssent);
    __syncthreads();
    float e = (j <= t) ? __expf(s - mx) : 0.f;
    red[j] = e; __syncthreads();
    for (int off = 64; off; off >>= 1) {
        if (j < off) red[j] += red[j + off];
        __syncthreads();
    }
    float esent = __expf(ssent - mx);
    float denom = red[0] + esent;
    a[base + j] = e / denom;
    if (j == 0) gsent[r] = esent / denom;
}

// per (t,b): LDS vocab scatter (ds_add_f32), fused vocab softmax + log mixture.
// logits live in out0 (logp region) and are overwritten in place.
__global__ __launch_bounds__(256) void final_kernel(
    const int* __restrict__ ids, const float* __restrict__ a,
    const float* __restrict__ gsent,
    float* __restrict__ out0, float* __restrict__ out1)
{
    __shared__ float pptr[10000];
    __shared__ float red[256];
    int r = blockIdx.x, t = r >> 5, b = r & 31;
    int tid = threadIdx.x;
    for (int v = tid; v < 10000; v += 256) pptr[v] = 0.f;
    __syncthreads();
    for (int j = tid; j <= t; j += 256) {
        float av = a[(long long)b * 16384 + t * 128 + j];
        int tok = ids[j * 32 + b];
        atomicAdd(&pptr[tok], av);
    }
    __syncthreads();
    float* logits = out0 + (long long)r * 10000;
    float lm = -INFINITY;
    for (int v = tid; v < 10000; v += 256) lm = fmaxf(lm, logits[v]);
    red[tid] = lm; __syncthreads();
    for (int off = 128; off; off >>= 1) {
        if (tid < off) red[tid] = fmaxf(red[tid], red[tid + off]);
        __syncthreads();
    }
    float mx = red[0];
    __syncthreads();
    float ls = 0.f;
    for (int v = tid; v < 10000; v += 256) ls += __expf(logits[v] - mx);
    red[tid] = ls; __syncthreads();
    for (int off = 128; off; off >>= 1) {
        if (tid < off) red[tid] += red[tid + off];
        __syncthreads();
    }
    float inv = 1.f / red[0];
    float g = gsent[r];
    for (int v = tid; v < 10000; v += 256) {
        float pv = __expf(logits[v] - mx) * inv;
        float pp = pptr[v];
        out0[(long long)r * 10000 + v] = __logf(pp + pv * g);
        out1[(long long)r * 10000 + v] = __logf(pp + g);
    }
}

// ---------------- host orchestration ---------------------------------------
extern "C" void kernel_launch(void* const* d_in, const int* in_sizes, int n_in,
                              void* d_out, int out_size, void* d_ws, size_t ws_size,
                              hipStream_t stream)
{
    (void)in_sizes; (void)n_in; (void)out_size; (void)ws_size;
    const int L = 128, B = 32, V = 10000, D = 512, G = 4 * D, M = L * B;

    const int*   ids      = (const int*)  d_in[0];
    const float* embed_w  = (const float*)d_in[1];
    const float* W_ih     = (const float*)d_in[2];
    const float* b_ih     = (const float*)d_in[3];
    const float* W_hh     = (const float*)d_in[4];
    const float* b_hh     = (const float*)d_in[5];
    const float* W1       = (const float*)d_in[6];
    const float* b1       = (const float*)d_in[7];
    const float* W2       = (const float*)d_in[8];
    const float* b2       = (const float*)d_in[9];
    const float* sentinel = (const float*)d_in[10];

    float* out0 = (float*)d_out;                 // logp  (also logits scratch)
    float* out1 = out0 + (long long)M * V;       // logptr

    char* w = (char*)d_ws;
    auto alloc = [&](size_t bytes) {
        char* p = w; w += (bytes + 255) & ~(size_t)255; return p;
    };
    _Float16* Wih16  = (_Float16*)alloc((size_t)G * D * 2);
    _Float16* Whh16  = (_Float16*)alloc((size_t)G * D * 2);
    _Float16* W1_16  = (_Float16*)alloc((size_t)V * D * 2);
    _Float16* W2_16  = (_Float16*)alloc((size_t)D * D * 2);
    _Float16* X16    = (_Float16*)alloc((size_t)M * D * 2);
    float*    gatesX = (float*)   alloc((size_t)M * G * 4);
    float*    gatesH = (float*)   alloc((size_t)B * G * 4);
    float*    c32    = (float*)   alloc((size_t)B * D * 4);
    _Float16* h16    = (_Float16*)alloc((size_t)B * D * 2);
    _Float16* H16    = (_Float16*)alloc((size_t)M * D * 2);
    _Float16* Q16    = (_Float16*)alloc((size_t)M * D * 2);
    float*    S      = (float*)   alloc((size_t)B * L * L * 4);
    float*    aAttn  = (float*)   alloc((size_t)B * L * L * 4);
    float*    sentB  = (float*)   alloc((size_t)M * 4);
    float*    gsentB = (float*)   alloc((size_t)M * 4);

    // weight casts + embedding gather
    cast_f32_f16<<<(G * D) / 256, 256, 0, stream>>>(W_ih, Wih16, G * D);
    cast_f32_f16<<<(G * D) / 256, 256, 0, stream>>>(W_hh, Whh16, G * D);
    cast_f32_f16<<<(V * D) / 256, 256, 0, stream>>>(W1, W1_16, V * D);
    cast_f32_f16<<<(D * D) / 256, 256, 0, stream>>>(W2, W2_16, D * D);
    embed_kernel<<<(M * D) / 256, 256, 0, stream>>>(ids, embed_w, X16);

    // hoisted input-gate GEMM: gatesX = X @ W_ih^T + b_ih + b_hh   (WMMA)
    gemm_wmma<<<dim3(G / 64, M / 128, 1), 256, 0, stream>>>(
        X16, Wih16, b_ih, b_hh, gatesX, nullptr,
        M, G, D, D, D, G, 0, 0, 0, 0);

    // LSTM recurrence: 128 sequential (matmul, update) pairs
    (void)hipMemsetAsync(h16, 0, (size_t)B * D * 2, stream);
    (void)hipMemsetAsync(c32, 0, (size_t)B * D * 4, stream);
    for (int t = 0; t < L; t++) {
        lstm_step_matmul<<<32, 256, 0, stream>>>(h16, Whh16, gatesX, gatesH, t);
        lstm_update<<<(B * D) / 256, 256, 0, stream>>>(gatesH, c32, h16, H16, t);
    }

    // Q = tanh(H @ W2^T + b2)  -> f16
    gemm_wmma<<<dim3(D / 64, M / 128, 1), 256, 0, stream>>>(
        H16, W2_16, b2, nullptr, nullptr, Q16,
        M, D, D, D, D, D, 0, 0, 0, 1);

    // sent = Q . sentinel
    sent_kernel<<<M / 8, 256, 0, stream>>>(Q16, sentinel, sentB);

    // batched scores: S[b] = Q_b(128x512) @ H_b^T  (strided batch, no transposes)
    gemm_wmma<<<dim3(L / 64, 1, B), 256, 0, stream>>>(
        Q16, H16, nullptr, nullptr, S, nullptr,
        L, L, D, B * D, B * D, L,
        (long long)D, (long long)D, (long long)L * L, 0);

    // causal softmax with sentinel
    attn_softmax<<<M, 128, 0, stream>>>(S, sentB, aAttn, gsentB);

    // logits = H @ W1^T + b1  -> written into out0 (logp region, in-place later)
    gemm_wmma<<<dim3((V + 63) / 64, M / 128, 1), 256, 0, stream>>>(
        H16, W1_16, b1, nullptr, out0, nullptr,
        M, V, D, D, D, V, 0, 0, 0, 0);

    // scatter + vocab softmax + log mixture
    final_kernel<<<M, 256, 0, stream>>>(ids, aAttn, gsentB, out0, out1);
}
